// BarnesHutKernel_72739566125069
// MI455X (gfx1250) — matrix operand order repaired and checked by reading, not compiled, not run on hardware
//
#include <hip/hip_runtime.h>
#include <cmath>

// CDNA5 / gfx1250 vector types for WMMA
typedef __attribute__((ext_vector_type(16))) _Float16 v16h;
typedef __attribute__((ext_vector_type(8)))  _Float16 v8h;
typedef __attribute__((ext_vector_type(8)))  float    v8f;

#define TT 2048   // targets
#define SS 1024   // sources
#define NW 8      // waves per block (wave32 -> 256 threads)

__device__ __forceinline__ float fast_silu(float x) {
  // x * rcp(1 + exp(-x)) : v_exp_f32 + v_add + v_rcp_f32 + v_mul
  return x * __builtin_amdgcn_rcpf(1.0f + __expf(-x));
}
__device__ __forceinline__ float fast_tanh(float x) {
  // 1 - 2/(exp(2x)+1); correct limits at +/-inf
  return 1.0f - 2.0f * __builtin_amdgcn_rcpf(__expf(2.0f * x) + 1.0f);
}

__global__ __launch_bounds__(256) void bh_mlp_wmma(
    const float* __restrict__ refLen,
    const float* __restrict__ srcP,   // (S,3)
    const float* __restrict__ tgtP,   // (T,3)
    const float* __restrict__ srcW,   // (S,)
    const float* __restrict__ srcA,   // (S,)
    const float* __restrict__ srcN,   // (S,3)
    const float* __restrict__ W1,     // (7,64) row-major
    const float* __restrict__ B1,     // (64,)
    const float* __restrict__ W2,     // (64,4) row-major
    const float* __restrict__ B2,     // (4,)
    float* __restrict__ out)          // (T,4)
{
  // Per-wave LDS slices (no cross-wave sharing -> no barriers needed;
  // LDS ops from the same wave are in-order on CDNA5).
  // s_feats16: rows 0..31 = live features (f16, 8 halves = 16B/row, one
  // ds_store_b128 per lane); rows 32..63 stay zero so the upper half-wave of
  // an A-tile (which must contribute K=16..31 = zeros) just reads row+32.
  __shared__ __align__(16) _Float16 s_feats16[NW][64][8];
  __shared__ __align__(16) _Float16 s_h[NW][16][72];   // h tile, row-major, padded stride
  __shared__ float                  s_out[NW][32][4];  // MLP outputs per pair

  const int tid   = threadIdx.x;
  const int w     = tid >> 5;       // wave id in block
  const int lane  = tid & 31;       // wave32 lane
  const int coll  = lane & 15;      // column / half-wave row index
  const bool upper = lane >= 16;
  const int t = blockIdx.x * NW + w;  // one target per wave

  // zero the pad rows once (one 16B store per lane)
  {
    v8h z;
    #pragma unroll
    for (int i = 0; i < 8; ++i) z[i] = (_Float16)0.0f;
    *reinterpret_cast<v8h*>(&s_feats16[w][32 + lane][0]) = z;
  }

  const float invL = __builtin_amdgcn_rcpf(refLen[0]);
  const float tx = tgtP[t*3+0], ty = tgtP[t*3+1], tz = tgtP[t*3+2];
  const float EPS2 = 1e-16f;

  // ---- Preload W1/W2 in WMMA B layout (f16), biases folded into C later ----
  // w1b B (K=32 x N=16, this code's contiguous order): lanes 0-15 hold K=0..15 in the 16 half slots,
  // lanes 16-31 hold K=16..31; N = lane&15.
  float b1v[4];
  #pragma unroll
  for (int nt = 0; nt < 4; ++nt) b1v[nt] = B1[nt*16 + coll];
  const float b2v = (coll < 4) ? B2[coll] : 0.0f;

  v16h w1b[4];   // K padded 7 -> 32 with zeros
  #pragma unroll
  for (int nt = 0; nt < 4; ++nt) {
    #pragma unroll
    for (int i = 0; i < 16; ++i) {
      const int k = (upper ? 16 : 0) + i;
      const float v = (k < 7) ? W1[k*64 + nt*16 + coll] : 0.0f;
      w1b[nt][i] = (_Float16)v;
    }
  }
  v16h w2b[2];   // K=64 as two K=32 chunks; N padded 4 -> 16 with zeros
  #pragma unroll
  for (int ch = 0; ch < 2; ++ch) {
    #pragma unroll
    for (int i = 0; i < 16; ++i) {
      const int k = 32*ch + (upper ? 8 : 0) + (i < 8 ? i : i + 8);
      const float v = (coll < 4) ? W2[k*4 + coll] : 0.0f;
      w2b[ch][i] = (_Float16)v;
    }
  }

  float acc0 = 0.f, accx = 0.f, accy = 0.f, accz = 0.f;

  for (int sb = 0; sb < SS; sb += 32) {
    const int s = sb + lane;             // pair handled by this lane

    // ---- geometry + features (fp32 VALU; co-executes with WMMA pipe) ----
    const float rx = (tx - srcP[s*3+0]) * invL;
    const float ry = (ty - srcP[s*3+1]) * invL;
    const float rz = (tz - srcP[s*3+2]) * invL;
    const float nx = srcN[s*3+0], ny = srcN[s*3+1], nz = srcN[s*3+2];
    const float r2 = rx*rx + ry*ry + rz*rz + EPS2;
    const float n2 = nx*nx + ny*ny + nz*nz + EPS2;
    const float inv_r = __builtin_amdgcn_rsqf(r2);
    const float inv_n = __builtin_amdgcn_rsqf(n2);
    const float log_r = 0.5f * __logf(r2);
    const float log_n = 0.5f * __logf(n2);
    const float cosv  = (rx*nx + ry*ny + rz*nz) * inv_r * inv_n;
    const float c2v   = cosv * cosv;
    const float p1 = cosv;
    const float p2 = 0.5f  * (3.0f*c2v - 1.0f);
    const float p3 = 0.5f  * (5.0f*c2v - 3.0f) * cosv;
    const float p4 = 0.125f* ((35.0f*c2v - 30.0f)*c2v + 3.0f);
    const float area = srcA[s];
    const float ws   = srcW[s];
    const float rhx = rx*inv_r, rhy = ry*inv_r, rhz = rz*inv_r;
    const float nhx = nx*inv_n, nhy = ny*inv_n, nhz = nz*inv_n;

    // pack the 7 features (+ zero pad) and store as one ds_store_b128
    {
      v8h f;
      f[0] = (_Float16)area;  f[1] = (_Float16)log_r; f[2] = (_Float16)log_n;
      f[3] = (_Float16)p1;    f[4] = (_Float16)p2;    f[5] = (_Float16)p3;
      f[6] = (_Float16)p4;    f[7] = (_Float16)0.0f;
      *reinterpret_cast<v8h*>(&s_feats16[w][lane][0]) = f;
    }

    if (sb + 32 < SS) __builtin_prefetch(&srcP[(s + 32) * 3], 0, 0);

    #pragma unroll
    for (int tile = 0; tile < 2; ++tile) {
      // ---- build A1 (16x32 f16; K padded 7->32). Lanes 0-15 hold K=0..7 in
      // halves 0-7 (row = lane&15); lanes>=16 hold K=16..31 => read zero row.
      const int arow = tile*16 + coll + (upper ? 32 : 0);
      const v8h f = *reinterpret_cast<const v8h*>(&s_feats16[w][arow][0]);
      v16h a1;
      #pragma unroll
      for (int i = 0; i < 8; ++i) { a1[i] = f[i]; a1[i+8] = (_Float16)0.0f; }

      // ---- GEMM1: h = silu(feats @ W1 + b1), N=64 as four 16-wide tiles ----
      #pragma unroll
      for (int nt = 0; nt < 4; ++nt) {
        v8f c;
        #pragma unroll
        for (int j = 0; j < 8; ++j) c[j] = b1v[nt];   // bias folded into C
        c = __builtin_amdgcn_wmma_f32_16x16x32_f16(
                false, a1, false, w1b[nt], (short)0, c, false, false);
        #pragma unroll
        for (int j = 0; j < 8; ++j) {
          const float hs = fast_silu(c[j]);
          const int row = j + (upper ? 8 : 0);        // C/D layout row
          s_h[w][row][nt*16 + coll] = (_Float16)hs;
        }
      }

      // ---- GEMM2: out = h @ W2 + b2 ; K=64 as two K=32 WMMAs, N padded 4->16.
      // A2 layout reads from row-major LDS: lane<16 base K=0(+chunk), lane>=16 base K=8;
      // halves 0-7 = [base..base+7], halves 8-15 = [base+16..base+23]  (16B aligned).
      v8f c2;
      #pragma unroll
      for (int j = 0; j < 8; ++j) c2[j] = b2v;
      #pragma unroll
      for (int ch = 0; ch < 2; ++ch) {
        const int kb = 32*ch + (upper ? 8 : 0);
        const v8h vlo = *reinterpret_cast<const v8h*>(&s_h[w][coll][kb]);
        const v8h vhi = *reinterpret_cast<const v8h*>(&s_h[w][coll][kb + 16]);
        v16h a2;
        #pragma unroll
        for (int i = 0; i < 8; ++i) { a2[i] = vlo[i]; a2[i+8] = vhi[i]; }
        c2 = __builtin_amdgcn_wmma_f32_16x16x32_f16(
                 false, a2, false, w2b[ch], (short)0, c2, false, false);
      }

      // scatter the 4 real output columns to LDS for per-pair postproc
      if (coll < 4) {
        #pragma unroll
        for (int j = 0; j < 8; ++j) {
          const int row = tile*16 + j + (upper ? 8 : 0);
          s_out[w][row][coll] = c2[j];
        }
      }
    }

    // ---- per-pair postprocessing (pair = lane; geometry still in registers) ----
    const float o0 = fast_tanh(s_out[w][lane][0]);
    const float o1 = fast_tanh(s_out[w][lane][1]);
    const float o2 = fast_tanh(s_out[w][lane][2]);
    const float o3 = fast_tanh(s_out[w][lane][3]);
    acc0 += ws * inv_r * o0;                           // decay_s = 1/r (D=3)
    const float crx = rhy*nhz - rhz*nhy;
    const float cry = rhz*nhx - rhx*nhz;
    const float crz = rhx*nhy - rhy*nhx;
    const float dv  = ws * inv_r * inv_r;              // decay_v = 1/r^2
    accx += dv * (o1*rhx + o2*nhx + o3*crx);
    accy += dv * (o1*rhy + o2*nhy + o3*cry);
    accz += dv * (o1*rhz + o2*nhz + o3*crz);
  }

  // ---- wave32 reduction over the 32 source-lanes ----
  #pragma unroll
  for (int off = 16; off > 0; off >>= 1) {
    acc0 += __shfl_xor(acc0, off, 32);
    accx += __shfl_xor(accx, off, 32);
    accy += __shfl_xor(accy, off, 32);
    accz += __shfl_xor(accz, off, 32);
  }
  if (lane == 0) {
    out[t*4+0] = acc0;
    out[t*4+1] = accx;
    out[t*4+2] = accy;
    out[t*4+3] = accz;
  }
}

extern "C" void kernel_launch(void* const* d_in, const int* in_sizes, int n_in,
                              void* d_out, int out_size, void* d_ws, size_t ws_size,
                              hipStream_t stream) {
  (void)in_sizes; (void)n_in; (void)out_size; (void)d_ws; (void)ws_size;
  const float* refLen = (const float*)d_in[0];
  const float* srcP   = (const float*)d_in[1];
  const float* tgtP   = (const float*)d_in[2];
  const float* srcW   = (const float*)d_in[3];
  const float* srcA   = (const float*)d_in[4];
  const float* srcN   = (const float*)d_in[5];
  const float* W1     = (const float*)d_in[6];
  const float* B1     = (const float*)d_in[7];
  const float* W2     = (const float*)d_in[8];
  const float* B2     = (const float*)d_in[9];
  float* out = (float*)d_out;

  dim3 grid(TT / NW);     // 256 blocks
  dim3 block(32 * NW);    // 256 threads = 8 wave32 waves
  hipLaunchKernelGGL(bh_mlp_wmma, grid, block, 0, stream,
                     refLen, srcP, tgtP, srcW, srcA, srcN, W1, B1, W2, B2, out);
}
